// Multi_Attention2D_73830487818429
// MI455X (gfx1250) — matrix operand order repaired and checked by reading, not compiled, not run on hardware
//
#include <hip/hip_runtime.h>
#include <hip/hip_bf16.h>

// ---------------------------------------------------------------------------
// Types for CDNA5 WMMA (wave32): v_wmma_f32_16x16x32_bf16
// ---------------------------------------------------------------------------
typedef __attribute__((ext_vector_type(16))) __bf16 v16bf;
typedef __attribute__((ext_vector_type(8)))  __bf16 v8bf;
typedef __attribute__((ext_vector_type(8)))  float  v8f;

static __device__ __forceinline__ v8f wmma_bf16(v16bf a, v16bf b, v8f c) {
  // 8 args: (neg_a, A, neg_b, B, c_mod, C, reuse_a, reuse_b)
  return __builtin_amdgcn_wmma_f32_16x16x32_bf16(false, a, false, b, (short)0, c,
                                                 false, false);
}

// A fragment (16x32 bf16, row-major source, row stride in elements).
// Lane L (L<16): row M=L, elements 0..7 = K 0..7, 8..15 = K 16..23.
// Lane L>=16:   row M=L-16, elements = K 8..15 and K 24..31.
static __device__ __forceinline__ v16bf load_a_frag(const __bf16* tile, int stride,
                                                    int lane) {
  int r    = lane & 15;
  int koff = (lane & 16) ? 8 : 0;
  const v8bf* p0 = (const v8bf*)(tile + r * stride + koff);
  const v8bf* p1 = (const v8bf*)(tile + r * stride + koff + 16);
  v8bf x = *p0, y = *p1;
  v16bf f;
#pragma unroll
  for (int i = 0; i < 8; ++i) { f[i] = x[i]; f[i + 8] = y[i]; }
  return f;
}

// B fragment (32x16 bf16). Source is B^T stored row-major (rows = N columns of B).
// Lane L: column N = L&15; elements 0..7 = K ks..ks+7, 8..15 = K ks+16..ks+23, ks = (L>=16)?8:0.
static __device__ __forceinline__ v16bf load_b_frag(const __bf16* tileT, int stride,
                                                    int lane) {
  int n  = lane & 15;
  int ks = (lane & 16) ? 8 : 0;
  const v8bf* p0 = (const v8bf*)(tileT + n * stride + ks);
  v8bf x = p0[0], y = p0[2];
  v16bf f;
#pragma unroll
  for (int i = 0; i < 8; ++i) { f[i] = x[i]; f[i + 8] = y[i]; }
  return f;
}

// ---------------------------------------------------------------------------
// Problem constants
// ---------------------------------------------------------------------------
#define BATCH 2
#define TLEN  1024
#define NIMG  (BATCH * TLEN)   // 2048 images
#define HH    16
#define WW    16
#define CIN   32
#define NHEADS 4
#define KDIM  16
#define KERND 64                // NHEADS*KDIM
#define DATT  4096              // H*W*KDIM
#define NBH   (NHEADS * BATCH)  // 8 batch-heads

// ---------------------------------------------------------------------------
// Kernel 1: 3x3 SAME conv (Cin=32 -> Cout=64) as implicit GEMM with WMMA.
// One image (16x16) per workgroup, 8 waves, each wave computes 8 16x16 tiles.
// mode 0: write qh/kh layout [head*B+b][t][pos*16+d] (bf16)
// mode 2: write vT layout    [head*B+b][pos*16+d][t] (bf16, V transposed)
// ---------------------------------------------------------------------------
__global__ void __launch_bounds__(256)
conv_qkv_kernel(const float* __restrict__ in,   // (2048,16,16,32) f32
                const float* __restrict__ wgt,  // (3,3,32,64) f32
                __bf16* __restrict__ out, int mode) {
  __shared__ __align__(16) __bf16 in_lds[18 * 18 * CIN];     // haloed image
  __shared__ __align__(16) __bf16 w_lds[9 * KERND * CIN];    // W^T per tap

  const int tid = threadIdx.x;
  const int bt  = blockIdx.x;

  for (int i = tid; i < 18 * 18 * CIN; i += 256) in_lds[i] = (__bf16)0.0f;
  __syncthreads();

  const float* img = in + (size_t)bt * (HH * WW * CIN);
  for (int i = tid; i < HH * WW * CIN; i += 256) {
    int c = i & 31, x = (i >> 5) & 15, y = i >> 9;
    in_lds[((y + 1) * 18 + (x + 1)) * CIN + c] = (__bf16)img[i];
  }
  // weights: global [tap][cin][cout] -> LDS [tap][cout][cin] (= B^T rows)
  for (int i = tid; i < 9 * KERND * CIN; i += 256) {
    int cin = i & 31, cout = (i >> 5) & 63, tap = i >> 11;
    w_lds[i] = (__bf16)wgt[(tap * CIN + cin) * KERND + cout];
  }
  __syncthreads();

  const int lane = tid & 31;
  const int wave = tid >> 5;
  const int bimg = bt >> 10;     // image batch index (T=1024)
  const int t    = bt & 1023;

  for (int j = 0; j < 8; ++j) {
    const int tile  = wave * 8 + j;
    const int mtile = tile >> 2;  // h row of the image
    const int ntile = tile & 3;   // == head (cout block of 16)
    v8f acc = {};
#pragma unroll
    for (int ky = 0; ky < 3; ++ky)
#pragma unroll
      for (int kx = 0; kx < 3; ++kx) {
        const __bf16* abase = in_lds + ((mtile + ky) * 18 + kx) * CIN;
        const __bf16* bbase = w_lds + ((ky * 3 + kx) * KERND + ntile * 16) * CIN;
        v16bf a = load_a_frag(abase, CIN, lane);
        v16bf b = load_b_frag(bbase, CIN, lane);
        acc = wmma_bf16(a, b, acc);
      }
    const int col   = lane & 15;          // d within head
    const int rbase = (lane >> 4) << 3;   // M offset 0 or 8
#pragma unroll
    for (int e = 0; e < 8; ++e) {
      int row = rbase + e;                       // w coordinate
      int pos = mtile * 16 + row;                // h*16 + w
      size_t bh = (size_t)(ntile * BATCH + bimg);
      size_t idx;
      if (mode == 2)
        idx = (bh * DATT + (size_t)pos * 16 + col) * TLEN + t;       // vT
      else
        idx = (bh * TLEN + t) * DATT + (size_t)pos * 16 + col;       // qh/kh
      out[idx] = (__bf16)acc[e];
    }
  }
}

// ---------------------------------------------------------------------------
// Kernel 2: S = (Q K^T) / 8 per batch-head. M=N=1024, K=4096, bf16 WMMA.
// Block tile 128x128, LDS-staged K steps of 32. Grid (8, 8, 8).
// ---------------------------------------------------------------------------
__global__ void __launch_bounds__(256)
qk_kernel(const __bf16* __restrict__ qh, const __bf16* __restrict__ kh,
          float* __restrict__ S) {
  __shared__ __align__(16) __bf16 At[128 * 32];
  __shared__ __align__(16) __bf16 Bt[128 * 32];

  const int tid = threadIdx.x, lane = tid & 31, wave = tid >> 5;
  const int bh = blockIdx.z;
  const int M0 = blockIdx.y * 128, N0 = blockIdx.x * 128;
  const __bf16* Q = qh + (size_t)bh * TLEN * DATT;
  const __bf16* K = kh + (size_t)bh * TLEN * DATT;

  const int wm = wave >> 1, wn = wave & 1;   // 4x2 wave grid -> 32x64 per wave
  v8f acc[2][4];
#pragma unroll
  for (int i = 0; i < 2; ++i)
#pragma unroll
    for (int j = 0; j < 4; ++j) acc[i][j] = {};

  const int chunk = tid & 3, r0 = tid >> 2;  // 64 rows per pass, 4x8 cols
  for (int kb = 0; kb < DATT; kb += 32) {
#pragma unroll
    for (int p = 0; p < 2; ++p) {
      int r = r0 + p * 64;
      *(v8bf*)&At[r * 32 + chunk * 8] =
          *(const v8bf*)&Q[(size_t)(M0 + r) * DATT + kb + chunk * 8];
      *(v8bf*)&Bt[r * 32 + chunk * 8] =
          *(const v8bf*)&K[(size_t)(N0 + r) * DATT + kb + chunk * 8];
    }
    __syncthreads();
    v16bf a[2], b[4];
#pragma unroll
    for (int i = 0; i < 2; ++i) a[i] = load_a_frag(At + (wm * 32 + i * 16) * 32, 32, lane);
#pragma unroll
    for (int j = 0; j < 4; ++j) b[j] = load_b_frag(Bt + (wn * 64 + j * 16) * 32, 32, lane);
#pragma unroll
    for (int i = 0; i < 2; ++i)
#pragma unroll
      for (int j = 0; j < 4; ++j) acc[i][j] = wmma_bf16(a[i], b[j], acc[i][j]);
    __syncthreads();
  }

  const int col = lane & 15, rbase = (lane >> 4) << 3;
#pragma unroll
  for (int i = 0; i < 2; ++i)
#pragma unroll
    for (int j = 0; j < 4; ++j)
#pragma unroll
      for (int e = 0; e < 8; ++e) {
        int row = M0 + wm * 32 + i * 16 + rbase + e;
        int cc  = N0 + wn * 64 + j * 16 + col;
        S[((size_t)bh * TLEN + row) * TLEN + cc] = acc[i][j][e] * 0.125f;
      }
}

// ---------------------------------------------------------------------------
// Kernel 3: row softmax over S (8192 rows of 1024), emits bf16 P.
// ---------------------------------------------------------------------------
__global__ void __launch_bounds__(256)
softmax_kernel(const float* __restrict__ S, __bf16* __restrict__ P) {
  __shared__ float red[256];
  const int r = blockIdx.x, tid = threadIdx.x;
  const float* row = S + (size_t)r * TLEN;

  float v[4];
  float mx = -3.0e38f;
#pragma unroll
  for (int j = 0; j < 4; ++j) { v[j] = row[tid + j * 256]; mx = fmaxf(mx, v[j]); }
  red[tid] = mx;
  __syncthreads();
  for (int s = 128; s > 0; s >>= 1) {
    if (tid < s) red[tid] = fmaxf(red[tid], red[tid + s]);
    __syncthreads();
  }
  mx = red[0];
  __syncthreads();

  float sum = 0.0f;
#pragma unroll
  for (int j = 0; j < 4; ++j) { v[j] = __expf(v[j] - mx); sum += v[j]; }
  red[tid] = sum;
  __syncthreads();
  for (int s = 128; s > 0; s >>= 1) {
    if (tid < s) red[tid] += red[tid + s];
    __syncthreads();
  }
  const float inv = 1.0f / red[0];
#pragma unroll
  for (int j = 0; j < 4; ++j)
    P[(size_t)r * TLEN + tid + j * 256] = (__bf16)(v[j] * inv);
}

// ---------------------------------------------------------------------------
// Kernel 4: Z = P @ V per batch-head. M=1024, K=1024, N=4096.
// B^T = vT rows (contiguous). Epilogue scatters into conv layout c' = d*4+head.
// Grid (32, 8, 8).
// ---------------------------------------------------------------------------
__global__ void __launch_bounds__(256)
pv_kernel(const __bf16* __restrict__ P, const __bf16* __restrict__ vT,
          __bf16* __restrict__ zc) {
  __shared__ __align__(16) __bf16 At[128 * 32];
  __shared__ __align__(16) __bf16 Bt[128 * 32];

  const int tid = threadIdx.x, lane = tid & 31, wave = tid >> 5;
  const int bh = blockIdx.z;
  const int M0 = blockIdx.y * 128, N0 = blockIdx.x * 128;
  const __bf16* A  = P + (size_t)bh * TLEN * TLEN;
  const __bf16* Vt = vT + (size_t)bh * DATT * TLEN;

  const int wm = wave >> 1, wn = wave & 1;
  v8f acc[2][4];
#pragma unroll
  for (int i = 0; i < 2; ++i)
#pragma unroll
    for (int j = 0; j < 4; ++j) acc[i][j] = {};

  const int chunk = tid & 3, r0 = tid >> 2;
  for (int kb = 0; kb < TLEN; kb += 32) {
#pragma unroll
    for (int p = 0; p < 2; ++p) {
      int r = r0 + p * 64;
      *(v8bf*)&At[r * 32 + chunk * 8] =
          *(const v8bf*)&A[(size_t)(M0 + r) * TLEN + kb + chunk * 8];
      *(v8bf*)&Bt[r * 32 + chunk * 8] =
          *(const v8bf*)&Vt[(size_t)(N0 + r) * TLEN + kb + chunk * 8];
    }
    __syncthreads();
    v16bf a[2], b[4];
#pragma unroll
    for (int i = 0; i < 2; ++i) a[i] = load_a_frag(At + (wm * 32 + i * 16) * 32, 32, lane);
#pragma unroll
    for (int j = 0; j < 4; ++j) b[j] = load_b_frag(Bt + (wn * 64 + j * 16) * 32, 32, lane);
#pragma unroll
    for (int i = 0; i < 2; ++i)
#pragma unroll
      for (int j = 0; j < 4; ++j) acc[i][j] = wmma_bf16(a[i], b[j], acc[i][j]);
    __syncthreads();
  }

  const int col = lane & 15, rbase = (lane >> 4) << 3;
  const int head = bh >> 1, bimg = bh & 1;
#pragma unroll
  for (int i = 0; i < 2; ++i)
#pragma unroll
    for (int j = 0; j < 4; ++j)
#pragma unroll
      for (int e = 0; e < 8; ++e) {
        int t = M0 + wm * 32 + i * 16 + rbase + e;
        int n = N0 + wn * 64 + j * 16 + col;   // pos*16 + d
        int pos = n >> 4, d = n & 15;
        size_t bt = (size_t)bimg * TLEN + t;
        zc[(bt * 256 + pos) * KERND + d * NHEADS + head] = (__bf16)acc[i][j][e];
      }
}

// ---------------------------------------------------------------------------
// Kernel 5: 3x3 SAME conv (Cin=64 -> Cout=32), fp32 output in reference layout.
// ---------------------------------------------------------------------------
__global__ void __launch_bounds__(256)
out_conv_kernel(const __bf16* __restrict__ zc,  // (2048,16,16,64) bf16
                const float* __restrict__ wgt,  // (3,3,64,32) f32
                float* __restrict__ out) {      // (2048,16,16,32) f32
  __shared__ __align__(16) __bf16 in_lds[18 * 18 * KERND];
  __shared__ __align__(16) __bf16 w_lds[9 * 32 * KERND];

  const int tid = threadIdx.x;
  const int bt  = blockIdx.x;

  for (int i = tid; i < 18 * 18 * KERND; i += 256) in_lds[i] = (__bf16)0.0f;
  __syncthreads();

  const __bf16* img = zc + (size_t)bt * (HH * WW * KERND);
  for (int i = tid; i < HH * WW * KERND; i += 256) {
    int c = i & 63, x = (i >> 6) & 15, y = i >> 10;
    in_lds[((y + 1) * 18 + (x + 1)) * KERND + c] = img[i];
  }
  // weights: global [tap][cin64][cout32] -> LDS [tap][cout][cin]
  for (int i = tid; i < 9 * 32 * KERND; i += 256) {
    int cin = i & 63, cout = (i >> 6) & 31, tap = i >> 11;
    w_lds[i] = (__bf16)wgt[(tap * KERND + cin) * 32 + cout];
  }
  __syncthreads();

  const int lane = tid & 31;
  const int wave = tid >> 5;

  for (int j = 0; j < 4; ++j) {
    const int tile  = wave * 4 + j;
    const int mtile = tile >> 1;   // h row
    const int ntile = tile & 1;    // cout block of 16
    v8f acc = {};
#pragma unroll
    for (int ky = 0; ky < 3; ++ky)
#pragma unroll
      for (int kx = 0; kx < 3; ++kx)
#pragma unroll
        for (int ks = 0; ks < KERND; ks += 32) {
          const __bf16* abase = in_lds + ((mtile + ky) * 18 + kx) * KERND + ks;
          const __bf16* bbase =
              w_lds + (ky * 3 + kx) * 32 * KERND + ntile * 16 * KERND + ks;
          v16bf a = load_a_frag(abase, KERND, lane);
          v16bf b = load_b_frag(bbase, KERND, lane);
          acc = wmma_bf16(a, b, acc);
        }
    const int col = lane & 15, rbase = (lane >> 4) << 3;
#pragma unroll
    for (int e = 0; e < 8; ++e) {
      int row = rbase + e;
      int pos = mtile * 16 + row;
      out[((size_t)bt * 256 + pos) * 32 + ntile * 16 + col] = acc[e];
    }
  }
}

// ---------------------------------------------------------------------------
// Host launcher
// ---------------------------------------------------------------------------
extern "C" void kernel_launch(void* const* d_in, const int* in_sizes, int n_in,
                              void* d_out, int out_size, void* d_ws, size_t ws_size,
                              hipStream_t stream) {
  const float* q  = (const float*)d_in[0];
  const float* k  = (const float*)d_in[1];
  const float* v  = (const float*)d_in[2];
  const float* qw = (const float*)d_in[3];
  const float* kw = (const float*)d_in[4];
  const float* vw = (const float*)d_in[5];
  const float* ow = (const float*)d_in[6];
  float* outp = (float*)d_out;

  char* ws = (char*)d_ws;
  const size_t QH_BYTES = (size_t)NBH * TLEN * DATT * sizeof(__bf16); // 64 MB
  const size_t S_BYTES  = (size_t)NBH * TLEN * TLEN * sizeof(float);  // 32 MB
  const size_t P_BYTES  = (size_t)NBH * TLEN * TLEN * sizeof(__bf16); // 16 MB

  __bf16* qh = (__bf16*)(ws);
  __bf16* kh = (__bf16*)(ws + QH_BYTES);
  __bf16* vT = (__bf16*)(ws + 2 * QH_BYTES);
  float*  S  = (float*) (ws + 3 * QH_BYTES);
  __bf16* P  = (__bf16*)(ws + 3 * QH_BYTES + S_BYTES);
  __bf16* zc = (__bf16*)(ws + 3 * QH_BYTES + S_BYTES + P_BYTES);

  conv_qkv_kernel<<<NIMG, 256, 0, stream>>>(q, qw, qh, 0);
  conv_qkv_kernel<<<NIMG, 256, 0, stream>>>(k, kw, kh, 0);
  conv_qkv_kernel<<<NIMG, 256, 0, stream>>>(v, vw, vT, 2);

  qk_kernel<<<dim3(8, 8, NBH), 256, 0, stream>>>(qh, kh, S);
  softmax_kernel<<<NBH * TLEN, 256, 0, stream>>>(S, P);
  pv_kernel<<<dim3(32, 8, NBH), 256, 0, stream>>>(P, vT, zc);

  out_conv_kernel<<<NIMG, 256, 0, stream>>>(zc, ow, outp);
}
